// DefaultAttention_27814208209133
// MI455X (gfx1250) — matrix operand order repaired and checked by reading, not compiled, not run on hardware
//
#include <hip/hip_runtime.h>
#include <hip/hip_bf16.h>

#define EMB   1024
#define SEQ   2048
#define NB    4
#define LDSK  72          // padded LDS row stride (bf16 elements): 64 data + 8 pad, 144B

typedef __attribute__((ext_vector_type(16))) __bf16 v16bf;
typedef __attribute__((ext_vector_type(8)))  __bf16 v8bf;
typedef __attribute__((ext_vector_type(8)))  float  v8f;
typedef __attribute__((ext_vector_type(4)))  unsigned int u32x4;
typedef __attribute__((ext_vector_type(8)))  int i32x8;
typedef __attribute__((ext_vector_type(4)))  int i32x4;

// ---------------------------------------------------------------------------
// Tensor Data Mover: load a (rows x 64) bf16 tile (row stride = stride_elems)
// from global into LDS, inserting 8 bf16 of padding after each 64-element row
// (pad_interval code 4 = 32 DWORDs, pad_amount code 3 = 4 DWORDs) so the LDS
// layout matches LDSK=72.  Issue from one wave; EXEC is ignored by TDM.
__device__ __forceinline__ void tdm_load_tile_bf16(const void* gsrc, void* lds_dst,
                                                   unsigned stride_elems, unsigned rows)
{
    unsigned long long ga = (unsigned long long)(uintptr_t)gsrc;
    unsigned lds = (unsigned)(uintptr_t)lds_dst;   // generic LDS addr low 32b = LDS offset

    u32x4 g0;
    g0.x = 1u;                                             // count=1, load, no gather
    g0.y = lds;                                            // lds_addr
    g0.z = (unsigned)(ga & 0xffffffffu);                   // global_addr[31:0]
    g0.w = (unsigned)((ga >> 32) & 0x01ffffffu) | (2u << 30); // global_addr[56:32], type=2

    i32x8 g1;
    g1[0] = (int)((1u << 16)     // data_size = 2 bytes
                | (1u << 20)     // pad_enable
                | (4u << 22)     // pad_interval: 32 DWORDs (= one 64-elem bf16 row)
                | (3u << 25));   // pad_amount : 4 DWORDs  (= 8 bf16)  -> LDSK = 72
    const unsigned td0 = 1u << 20, td1 = 1u << 20;         // generous dims: no OOB path
    g1[1] = (int)((td0 & 0xffffu) << 16);                  // tensor_dim0[15:0]
    g1[2] = (int)((td0 >> 16) | ((td1 & 0xffffu) << 16));  // tensor_dim0[31:16], dim1[15:0]
    g1[3] = (int)((td1 >> 16) | (64u << 16));              // tensor_dim1[31:16], tile_dim0=64
    g1[4] = (int)(rows & 0xffffu);                         // tile_dim1=rows, tile_dim2=0
    g1[5] = (int)stride_elems;                             // tensor_dim0_stride[31:0]
    g1[6] = 0;                                             // stride hi, dim1_stride lo
    g1[7] = 0;

    i32x4 gz = {0, 0, 0, 0};
#if __clang_major__ >= 23
    i32x8 gz8 = {0, 0, 0, 0, 0, 0, 0, 0};
    __builtin_amdgcn_tensor_load_to_lds(g0, g1, gz, gz, gz8, 0);
#else
    __builtin_amdgcn_tensor_load_to_lds(g0, g1, gz, gz, 0);
#endif
}

// ---------------------------------------------------------------------------
// Fragment loaders (wave32), ISA 7.12.2 layouts.  LDS tiles: [row][k], stride LDSK.
// A 16x32 bf16: lane m=lane&15; lanes<16: K runs {0..7,16..23}; lanes>=16: +8.
__device__ __forceinline__ v16bf load_frag_a(const __bf16* As, int row_off, int kk, int lane) {
    const int m  = lane & 15;
    const int hb = (lane >> 4) << 3;
    const __bf16* p = As + (row_off + m) * LDSK + kk + hb;
    v8bf lo = *(const v8bf*)(p);
    v8bf hi = *(const v8bf*)(p + 16);
    v16bf r;
#pragma unroll
    for (int i = 0; i < 8; ++i) { r[i] = lo[i]; r[i + 8] = hi[i]; }
    return r;
}

// B 32x16 bf16: lane n=lane&15 holds column n; lanes<16: K runs {0..7,16..23}; lanes>=16: +8.
// Tile stored [n][k] so each lane reads two runs of 8 contiguous bf16, starting 16 apart.
__device__ __forceinline__ v16bf load_frag_b(const __bf16* Bs, int col_off, int kk, int lane) {
    const int n  = lane & 15;
    const int kb = (lane >> 4) << 3;
    const __bf16* p = Bs + (col_off + n) * LDSK + kk + kb;
    v8bf lo = *(const v8bf*)(p);
    v8bf hi = *(const v8bf*)(p + 16);
    v16bf r;
#pragma unroll
    for (int i = 0; i < 8; ++i) { r[i] = lo[i]; r[i + 8] = hi[i]; }
    return r;
}

// ---------------------------------------------------------------------------
// Kernel 1: QKV projection.  Y[r][c] = sum_e X[r][e]*W[c][e] + b[c]
// block tile 128x64, 4 waves, each wave 32 rows x 64 cols (8 accumulators).
__global__ __launch_bounds__(128) void qkv_kernel(
    const float* __restrict__ x,
    const float* __restrict__ Wk, const float* __restrict__ bk,
    const float* __restrict__ Wq, const float* __restrict__ bq,
    const float* __restrict__ Wv, const float* __restrict__ bv,
    __bf16* __restrict__ Kb, __bf16* __restrict__ Qb, __bf16* __restrict__ Vb)
{
    const float* W;  const float* bi;  __bf16* out;
    if (blockIdx.z == 0)      { W = Wk; bi = bk; out = Kb; }
    else if (blockIdx.z == 1) { W = Wq; bi = bq; out = Qb; }
    else                      { W = Wv; bi = bv; out = Vb; }

    __shared__ __align__(16) __bf16 As[128 * LDSK];
    __shared__ __align__(16) __bf16 Bs[64 * LDSK];

    const int tid  = threadIdx.x;
    const int lane = tid & 31;
    const int wv   = tid >> 5;
    const int row0 = blockIdx.x * 128;
    const int col0 = blockIdx.y * 64;

    v8f acc[2][4];
#pragma unroll
    for (int u = 0; u < 2; ++u)
#pragma unroll
        for (int i = 0; i < 4; ++i) acc[u][i] = (v8f)0.0f;

    for (int k0 = 0; k0 < EMB; k0 += 64) {
        __syncthreads();
        for (int c = tid; c < 128 * 16; c += 128) {        // X: fp32 -> bf16
            int r = c >> 4, kc = (c & 15) << 2;
            float4 v = *(const float4*)(x + (size_t)(row0 + r) * EMB + k0 + kc);
            __bf16* dp = &As[r * LDSK + kc];
            dp[0] = (__bf16)v.x; dp[1] = (__bf16)v.y; dp[2] = (__bf16)v.z; dp[3] = (__bf16)v.w;
        }
        for (int c = tid; c < 64 * 16; c += 128) {         // W tile [col][e]
            int r = c >> 4, kc = (c & 15) << 2;
            float4 v = *(const float4*)(W + (size_t)(col0 + r) * EMB + k0 + kc);
            __bf16* dp = &Bs[r * LDSK + kc];
            dp[0] = (__bf16)v.x; dp[1] = (__bf16)v.y; dp[2] = (__bf16)v.z; dp[3] = (__bf16)v.w;
        }
        __syncthreads();
#pragma unroll
        for (int kk = 0; kk < 64; kk += 32) {
            v16bf a0 = load_frag_a(As, wv * 32,      kk, lane);
            v16bf a1 = load_frag_a(As, wv * 32 + 16, kk, lane);
#pragma unroll
            for (int nt = 0; nt < 4; ++nt) {
                v16bf b = load_frag_b(Bs, nt * 16, kk, lane);
                acc[0][nt] = __builtin_amdgcn_wmma_f32_16x16x32_bf16(
                    false, a0, false, b, (short)0, acc[0][nt], false, false);
                acc[1][nt] = __builtin_amdgcn_wmma_f32_16x16x32_bf16(
                    false, a1, false, b, (short)0, acc[1][nt], false, false);
            }
        }
    }

    const int n  = lane & 15;
    const int hb = (lane >> 4) << 3;
#pragma unroll
    for (int nt = 0; nt < 4; ++nt) {
        int   col  = col0 + nt * 16 + n;
        float bval = bi[col];
#pragma unroll
        for (int u = 0; u < 2; ++u)
#pragma unroll
            for (int r = 0; r < 8; ++r) {
                int row = row0 + wv * 32 + u * 16 + hb + r;
                out[(size_t)row * EMB + col] = (__bf16)(acc[u][nt][r] + bval);
            }
    }
}

// ---------------------------------------------------------------------------
// Kernel 2: S = K.Q^T/sqrt(E), causal.  Block tile 128 (s) x 64 (t).
// Double-buffered TDM pipeline: wave 0 issues next tile pair, then waits
// tensorcnt<=2 (in-order completion => current pair resident) while the
// next transfer overlaps compute.
__global__ __launch_bounds__(128) void scores_kernel(
    const __bf16* __restrict__ Kb, const __bf16* __restrict__ Qb,
    float* __restrict__ Ssc)
{
    const int i = blockIdx.x, j = blockIdx.y, b = blockIdx.z;
    if (j > 2 * i + 1) return;

    __shared__ __align__(16) __bf16 As[2][128 * LDSK];
    __shared__ __align__(16) __bf16 Bs[2][64 * LDSK];

    const int tid  = threadIdx.x;
    const int lane = tid & 31;
    const int wv   = tid >> 5;

    const __bf16* Kp = Kb + ((size_t)b * SEQ + i * 128) * EMB;
    const __bf16* Qp = Qb + ((size_t)b * SEQ + j * 64) * EMB;

    v8f acc[2][4];
#pragma unroll
    for (int u = 0; u < 2; ++u)
#pragma unroll
        for (int t = 0; t < 4; ++t) acc[u][t] = (v8f)0.0f;

    if (wv == 0) {                       // prologue: fill buffer 0
        tdm_load_tile_bf16(Kp, As[0], EMB, 128);
        tdm_load_tile_bf16(Qp, Bs[0], EMB, 64);
    }

    int ib = 0;
    for (int k0 = 0; k0 < EMB; k0 += 64, ib ^= 1) {
        __syncthreads();                 // everyone done with As[ib^1]/Bs[ib^1]
        if (wv == 0) {
            if (k0 + 64 < EMB) {
                tdm_load_tile_bf16(Kp + k0 + 64, As[ib ^ 1], EMB, 128);
                tdm_load_tile_bf16(Qp + k0 + 64, Bs[ib ^ 1], EMB, 64);
                __builtin_amdgcn_s_wait_tensorcnt(2);   // current pair resident
            } else {
                __builtin_amdgcn_s_wait_tensorcnt(0);   // last chunk
            }
        }
        __syncthreads();                 // release compute on buffer ib
#pragma unroll
        for (int kk = 0; kk < 64; kk += 32) {
            v16bf a0 = load_frag_a(As[ib], wv * 32,      kk, lane);
            v16bf a1 = load_frag_a(As[ib], wv * 32 + 16, kk, lane);
#pragma unroll
            for (int nt = 0; nt < 4; ++nt) {
                v16bf bb = load_frag_b(Bs[ib], nt * 16, kk, lane);
                acc[0][nt] = __builtin_amdgcn_wmma_f32_16x16x32_bf16(
                    false, a0, false, bb, (short)0, acc[0][nt], false, false);
                acc[1][nt] = __builtin_amdgcn_wmma_f32_16x16x32_bf16(
                    false, a1, false, bb, (short)0, acc[1][nt], false, false);
            }
        }
    }

    const float scale = 0.03125f;                  // 1/sqrt(1024)
    const int n  = lane & 15;
    const int hb = (lane >> 4) << 3;
#pragma unroll
    for (int nt = 0; nt < 4; ++nt) {
#pragma unroll
        for (int u = 0; u < 2; ++u)
#pragma unroll
            for (int r = 0; r < 8; ++r) {
                int s_g = i * 128 + wv * 32 + u * 16 + hb + r;
                int t_g = j * 64 + nt * 16 + n;
                float v = acc[u][nt][r] * scale;
                if (t_g > s_g) v = -__builtin_inff();
                Ssc[((size_t)b * SEQ + s_g) * SEQ + t_g] = v;
            }
    }
}

// ---------------------------------------------------------------------------
// Kernel 3: row softmax over t<=s; bf16 P, zeros for t>s.
__global__ __launch_bounds__(256) void softmax_kernel(
    const float* __restrict__ Ssc, __bf16* __restrict__ P)
{
    const int gid = blockIdx.x;
    const int b = gid >> 11;
    const int s = gid & (SEQ - 1);
    const float* row  = Ssc + ((size_t)b * SEQ + s) * SEQ;
    __bf16*      prow = P   + ((size_t)b * SEQ + s) * SEQ;

    __shared__ float red[256];
    const int tid = threadIdx.x;

    float m = -__builtin_inff();
    for (int t = tid; t <= s; t += 256) m = fmaxf(m, row[t]);
    red[tid] = m; __syncthreads();
#pragma unroll
    for (int off = 128; off > 0; off >>= 1) {
        if (tid < off) red[tid] = fmaxf(red[tid], red[tid + off]);
        __syncthreads();
    }
    m = red[0]; __syncthreads();

    float sum = 0.0f;
    for (int t = tid; t <= s; t += 256) sum += __expf(row[t] - m);
    red[tid] = sum; __syncthreads();
#pragma unroll
    for (int off = 128; off > 0; off >>= 1) {
        if (tid < off) red[tid] += red[tid + off];
        __syncthreads();
    }
    const float inv = 1.0f / red[0];

    for (int t = tid; t < SEQ; t += 256)
        prow[t] = (t <= s) ? (__bf16)(__expf(row[t] - m) * inv) : (__bf16)0.0f;
}

// ---------------------------------------------------------------------------
// Kernel 4: O = P.V  (block tile 128 rows x 64 d-cols; t-blocks 0..2i+1)
// P tile via double-buffered TDM; V tile transposed manually (overlaps the
// in-flight TDM since it runs between the two barriers).
__global__ __launch_bounds__(128) void pv_kernel(
    const __bf16* __restrict__ P, const __bf16* __restrict__ Vb,
    float* __restrict__ out)
{
    const int rb = blockIdx.x;           // 0..63
    const int b  = rb >> 4;
    const int il = rb & 15;
    const int n0 = blockIdx.y * 64;

    __shared__ __align__(16) __bf16 As[2][128 * LDSK];  // P tile [s][t]
    __shared__ __align__(16) __bf16 Bs[64 * LDSK];      // V^T tile [d][t]

    const int tid  = threadIdx.x;
    const int lane = tid & 31;
    const int wv   = tid >> 5;

    const __bf16* Pp = P  + ((size_t)b * SEQ + il * 128) * SEQ;
    const __bf16* Vp = Vb + (size_t)b * SEQ * EMB;

    v8f acc[2][4];
#pragma unroll
    for (int u = 0; u < 2; ++u)
#pragma unroll
        for (int t = 0; t < 4; ++t) acc[u][t] = (v8f)0.0f;

    const int t_end = il * 128 + 64;     // last t-block start = (2*il+1)*64
    if (wv == 0) tdm_load_tile_bf16(Pp, As[0], SEQ, 128);

    int ib = 0;
    for (int t0 = 0; t0 <= t_end; t0 += 64, ib ^= 1) {
        __syncthreads();                 // prev compute done; As[ib^1], Bs free
        if (wv == 0) {
            if (t0 + 64 <= t_end) {
                tdm_load_tile_bf16(Pp + t0 + 64, As[ib ^ 1], SEQ, 128);
                __builtin_amdgcn_s_wait_tensorcnt(1);
            } else {
                __builtin_amdgcn_s_wait_tensorcnt(0);
            }
        }
        // V transposed: Bs[d][t] = V[t0+t][n0+d]; coalesced global reads.
        for (int c = tid; c < 64 * 8; c += 128) {
            int tt = c >> 3, nc = (c & 7) << 3;
            uint4 raw = *(const uint4*)(Vp + (size_t)(t0 + tt) * EMB + n0 + nc);
            const __bf16* hv = (const __bf16*)&raw;
#pragma unroll
            for (int jj = 0; jj < 8; ++jj)
                Bs[(nc + jj) * LDSK + tt] = hv[jj];
            if (t0 + 64 <= t_end)
                __builtin_prefetch(Vp + (size_t)(t0 + 64 + tt) * EMB + n0 + nc, 0, 0);
        }
        __syncthreads();
#pragma unroll
        for (int kk = 0; kk < 64; kk += 32) {
            v16bf a0 = load_frag_a(As[ib], wv * 32,      kk, lane);
            v16bf a1 = load_frag_a(As[ib], wv * 32 + 16, kk, lane);
#pragma unroll
            for (int nt = 0; nt < 4; ++nt) {
                v16bf bb = load_frag_b(Bs, nt * 16, kk, lane);
                acc[0][nt] = __builtin_amdgcn_wmma_f32_16x16x32_bf16(
                    false, a0, false, bb, (short)0, acc[0][nt], false, false);
                acc[1][nt] = __builtin_amdgcn_wmma_f32_16x16x32_bf16(
                    false, a1, false, bb, (short)0, acc[1][nt], false, false);
            }
        }
    }

    const int n  = lane & 15;
    const int hb = (lane >> 4) << 3;
#pragma unroll
    for (int nt = 0; nt < 4; ++nt) {
#pragma unroll
        for (int u = 0; u < 2; ++u)
#pragma unroll
            for (int r = 0; r < 8; ++r) {
                int s = il * 128 + wv * 32 + u * 16 + hb + r;
                int d = n0 + nt * 16 + n;
                out[((size_t)b * SEQ + s) * EMB + d] = acc[u][nt][r];
            }
    }
}

// ---------------------------------------------------------------------------
extern "C" void kernel_launch(void* const* d_in, const int* in_sizes, int n_in,
                              void* d_out, int out_size, void* d_ws, size_t ws_size,
                              hipStream_t stream) {
    const float* x  = (const float*)d_in[0];
    const float* Wk = (const float*)d_in[1];
    const float* bk = (const float*)d_in[2];
    const float* Wq = (const float*)d_in[3];
    const float* bq = (const float*)d_in[4];
    const float* Wv = (const float*)d_in[5];
    const float* bv = (const float*)d_in[6];
    float* out = (float*)d_out;

    char* ws = (char*)d_ws;
    const size_t KQV_BYTES = (size_t)NB * SEQ * EMB * 2;   // 16 MB each
    const size_t S_BYTES   = (size_t)NB * SEQ * SEQ * 4;   // 64 MB
    __bf16* Kb  = (__bf16*)(ws);
    __bf16* Qb  = (__bf16*)(ws + KQV_BYTES);
    __bf16* Vb  = (__bf16*)(ws + 2 * KQV_BYTES);
    float*  Ssc = (float*) (ws + 3 * KQV_BYTES);
    __bf16* P   = (__bf16*)(ws + 3 * KQV_BYTES + S_BYTES);

    qkv_kernel    <<<dim3(64, 16, 3), 128, 0, stream>>>(x, Wk, bk, Wq, bq, Wv, bv, Kb, Qb, Vb);
    scores_kernel <<<dim3(16, 32, 4), 128, 0, stream>>>(Kb, Qb, Ssc);
    softmax_kernel<<<dim3(NB * SEQ),  256, 0, stream>>>(Ssc, P);
    pv_kernel     <<<dim3(64, 16),    128, 0, stream>>>(P, Vb, out);
}
